// MSSC_58514634441112
// MI455X (gfx1250) — matrix operand order repaired and checked by reading, not compiled, not run on hardware
//
#include <hip/hip_runtime.h>
#include <stdint.h>

// ---------------- problem constants ----------------
#define BATCH   2
#define NPTS    65536
#define MPTS    (BATCH*NPTS)     // 131072 points total
#define HID     32
#define NSCALE  8
#define CATDIM  (HID*NSCALE)     // 256
#define OUTDIM  128
#define NTAP    27
#define NTAPP   28               // padded taps per row (16B-aligned index rows)

// ---------------- WMMA vector types (gfx1250, wave32) ----------------
typedef __attribute__((ext_vector_type(16))) __bf16        v16bf;
typedef __attribute__((ext_vector_type(8)))  float         v8f;
typedef __attribute__((ext_vector_type(4)))  unsigned int  u32x4;
typedef __attribute__((ext_vector_type(4)))  float         f32x4;

union Frag16 {
    struct { u32x4 lo, hi; } q;   // two 16-byte halves = 16 bf16 values
    v16bf v;
};

// native f32<->bf16 converts (clang emits v_cvt_pk_bf16_f32 / v_cvt_f32_bf16)
__device__ __forceinline__ unsigned short f2bf(float f) {
    __bf16 h = (__bf16)f;
    return __builtin_bit_cast(unsigned short, h);
}
__device__ __forceinline__ float bf2f(unsigned short u) {
    return (float)__builtin_bit_cast(__bf16, u);
}
// pack two f32 -> packed bf16x2 (one v_cvt_pk_bf16_f32)
__device__ __forceinline__ unsigned int pack2bf(float lo, float hi) {
    return (unsigned int)f2bf(lo) | ((unsigned int)f2bf(hi) << 16);
}

// ---------------- helper kernels ----------------
__global__ void k_fill_i32(int* __restrict__ ptr, int val, int n) {
    int i = blockIdx.x * blockDim.x + threadIdx.x;
    if (i < n) ptr[i] = val;
}

// mlp_1: pts = p @ w1 + b1   (3 -> 32), store bf16 rows of 32
__global__ void k_pts(const float* __restrict__ p, const float* __restrict__ w1,
                      const float* __restrict__ b1, unsigned short* __restrict__ pts) {
    int m = blockIdx.x * blockDim.x + threadIdx.x;
    if (m >= MPTS) return;
    float x = p[m*3+0], y = p[m*3+1], z = p[m*3+2];
    #pragma unroll 8
    for (int h = 0; h < HID; h += 2) {
        float a0 = b1[h]   + x*w1[0*HID+h]   + y*w1[1*HID+h]   + z*w1[2*HID+h];
        float a1 = b1[h+1] + x*w1[0*HID+h+1] + y*w1[1*HID+h+1] + z*w1[2*HID+h+1];
        *(unsigned int*)(pts + m*HID + h) = pack2bf(a0, a1);
    }
}

// per-batch min of floor(p/g)
__global__ void k_gmin(const float* __restrict__ p, float g, int* __restrict__ minbuf) {
    int m = blockIdx.x * blockDim.x + threadIdx.x;
    if (m >= MPTS) return;
    int b = m / NPTS;
    #pragma unroll
    for (int d = 0; d < 3; ++d) {
        int c = (int)floorf(p[m*3+d] / g);
        atomicMin(&minbuf[b*3+d], c);
    }
}

// voxel hash: scatter-max of flat point id into grid; pack grid coords
__global__ void k_hash(const float* __restrict__ p, float g,
                       const int* __restrict__ minbuf, int D,
                       int* __restrict__ grid, int* __restrict__ gcpack) {
    int m = blockIdx.x * blockDim.x + threadIdx.x;
    if (m >= MPTS) return;
    int b = m / NPTS;
    int gx = (int)floorf(p[m*3+0] / g) - minbuf[b*3+0];
    int gy = (int)floorf(p[m*3+1] / g) - minbuf[b*3+1];
    int gz = (int)floorf(p[m*3+2] / g) - minbuf[b*3+2];
    gcpack[m] = gx | (gy << 10) | (gz << 20);
    atomicMax(&grid[((b*D + gx)*D + gy)*D + gz], m);
}

// neighbor table, ROW-MAJOR PADDED: nbrR[m*28 + k] = voxel rep id or -1.
// 28-int rows are 112B = 16B-aligned -> conv loads indices as u32x4 (4 taps/load).
__global__ void k_nbr(const int* __restrict__ gcpack, const int* __restrict__ grid,
                      int D, int* __restrict__ nbrR) {
    int m = blockIdx.x * blockDim.x + threadIdx.x;
    if (m >= MPTS) return;
    int b  = m / NPTS;
    int pk = gcpack[m];
    int gx = pk & 1023, gy = (pk >> 10) & 1023, gz = (pk >> 20) & 1023;
    #pragma unroll
    for (int k = 0; k < NTAP; ++k) {
        int v;
        if (k == 13) {
            v = m;  // center tap is always self
        } else {
            int nx = gx + (k/9) - 1;
            int ny = gy + ((k/3)%3) - 1;
            int nz = gz + (k%3) - 1;
            v = -1;
            if (nx >= 0 && nx < D && ny >= 0 && ny < D && nz >= 0 && nz < D)
                v = grid[((b*D + nx)*D + ny)*D + nz];
        }
        nbrR[(size_t)m*NTAPP + k] = v;
    }
    nbrR[(size_t)m*NTAPP + 27] = -1;   // pad entry (loaded, never used)
}

// repack nmat 32x32 f32 matrices into WMMA B-fragments with COLUMN PAIRING:
// tile t computes output columns N = 2*(lane&15) + t, so each lane's two
// results per row are adjacent -> packed bf16x2 stores in the epilogue.
__global__ void k_repack32(const float* __restrict__ src, unsigned short* __restrict__ dst, int nmat) {
    int idx = blockIdx.x * blockDim.x + threadIdx.x;
    if (idx >= nmat * 1024) return;
    int mat = idx >> 10, rem = idx & 1023;
    int t = rem >> 9, lane = (rem >> 4) & 31, j = rem & 15;
    int K  = ((j >> 3) << 4) + ((lane >> 4) << 3) + (j & 7);
    int Nc = 2*(lane & 15) + t;
    dst[idx] = f2bf(src[mat*1024 + K*HID + Nc]);
}

// repack w2 [256,128] into 8 K-chunks x 8 N-tiles with 4-WAY COLUMN GROUPING:
// tile t computes N = (t>>2)*64 + 4*(lane&15) + (t&3) -> float4 stores.
__global__ void k_repack_w2(const float* __restrict__ src, unsigned short* __restrict__ dst) {
    int idx = blockIdx.x * blockDim.x + threadIdx.x;
    if (idx >= 64 * 512) return;
    int frag = idx >> 9, rem = idx & 511;
    int kk = frag >> 3, t = frag & 7;
    int lane = rem >> 4, j = rem & 15;
    int K  = kk*32 + ((j >> 3) << 4) + ((lane >> 4) << 3) + (j & 7);
    int Nc = (t >> 2)*64 + 4*(lane & 15) + (t & 3);
    dst[idx] = f2bf(src[K*OUTDIM + Nc]);
}

// shared epilogue: D(+bias)(+residuals) -> packed bf16x2 b32 stores
// c0 holds even columns (2c), c1 holds odd columns (2c+1), c = lane&15
__device__ __forceinline__ void epi_pair(
        int tile, int lane, v8f c0, v8f c1,
        const float* __restrict__ bias,
        const unsigned short* __restrict__ res1, const unsigned short* __restrict__ res2,
        unsigned short* __restrict__ fout, int outStride) {
    int cp = 2*(lane & 15), kh = lane >> 4;
    float b0 = bias[cp], b1 = bias[cp+1];
    #pragma unroll
    for (int r = 0; r < 8; ++r) {
        int orow = tile*16 + r + kh*8;
        float v0 = c0[r] + b0, v1 = c1[r] + b1;
        if (res1) {
            unsigned int rr = *(const unsigned int*)(res1 + (size_t)orow*HID + cp);
            v0 += bf2f((unsigned short)(rr & 0xffff));
            v1 += bf2f((unsigned short)(rr >> 16));
        }
        if (res2) {
            unsigned int rr = *(const unsigned int*)(res2 + (size_t)orow*HID + cp);
            v0 += bf2f((unsigned short)(rr & 0xffff));
            v1 += bf2f((unsigned short)(rr >> 16));
        }
        *(unsigned int*)(fout + (size_t)orow*outStride + cp) = pack2bf(v0, v1);
    }
}

// ---------------- WMMA kernels ----------------
// per-scale MLP: h0 = pts @ Wmlp + b ; TWO tiles per wave -> 4 independent
// WMMA chains fill the bf16 WMMA hazard slots.
__global__ void __launch_bounds__(128)
k_mlp_wmma(const unsigned short* __restrict__ pts, const unsigned short* __restrict__ wpack,
           const float* __restrict__ bias, unsigned short* __restrict__ out) {
    int wave = (blockIdx.x * blockDim.x + threadIdx.x) >> 5;
    int lane = threadIdx.x & 31;
    int row = lane & 15, kh = lane >> 4;
    int tileA = wave*2, tileB = wave*2 + 1;

    const unsigned short* arA = pts + (size_t)(tileA*16 + row) * HID;
    const unsigned short* arB = pts + (size_t)(tileB*16 + row) * HID;
    Frag16 aA, aB, b0, b1;
    aA.q.lo = *(const u32x4*)(arA + kh*8);
    aA.q.hi = *(const u32x4*)(arA + 16 + kh*8);
    aB.q.lo = *(const u32x4*)(arB + kh*8);
    aB.q.hi = *(const u32x4*)(arB + 16 + kh*8);
    b0.q.lo = *(const u32x4*)(wpack + lane*16);
    b0.q.hi = *(const u32x4*)(wpack + lane*16 + 8);
    b1.q.lo = *(const u32x4*)(wpack + 512 + lane*16);
    b1.q.hi = *(const u32x4*)(wpack + 512 + lane*16 + 8);

    v8f cA0 = (v8f)0.0f, cA1 = (v8f)0.0f, cB0 = (v8f)0.0f, cB1 = (v8f)0.0f;
    cA0 = __builtin_amdgcn_wmma_f32_16x16x32_bf16(false, aA.v, false, b0.v, (short)0, cA0, false, false);
    cB0 = __builtin_amdgcn_wmma_f32_16x16x32_bf16(false, aB.v, false, b0.v, (short)0, cB0, false, false);
    cA1 = __builtin_amdgcn_wmma_f32_16x16x32_bf16(false, aA.v, false, b1.v, (short)0, cA1, false, false);
    cB1 = __builtin_amdgcn_wmma_f32_16x16x32_bf16(false, aB.v, false, b1.v, (short)0, cB1, false, false);

    epi_pair(tileA, lane, cA0, cA1, bias, nullptr, nullptr, out, HID);
    epi_pair(tileB, lane, cB0, cB1, bias, nullptr, nullptr, out, HID);
}

// submanifold conv: out = sum_k (fin[nbr_k] @ Wk) + bias + res1 (+ res2), bf16 out
// Block = 8 waves; weights staged in LDS (54 KB); TWO tiles per wave;
// neighbor indices consumed 4-taps-per-b128 from the padded row-major table.
__global__ void __launch_bounds__(256)
k_conv_wmma(const unsigned short* __restrict__ fin, const int* __restrict__ nbrR,
            const unsigned short* __restrict__ wpack, const float* __restrict__ bias,
            const unsigned short* __restrict__ res1, const unsigned short* __restrict__ res2,
            unsigned short* __restrict__ fout, int outStride) {
    __shared__ unsigned short sw[NTAP*1024];          // 27 taps x 2 frags x 512 halfs
    {
        u32x4*       d = (u32x4*)sw;
        const u32x4* s = (const u32x4*)wpack;
        for (int i = threadIdx.x; i < NTAP*128; i += 256) d[i] = s[i];
    }
    __syncthreads();

    int waveid = threadIdx.x >> 5;
    int lane   = threadIdx.x & 31;
    int row = lane & 15, kh = lane >> 4;
    int tileA = blockIdx.x*16 + waveid;               // block covers 16 tiles
    int tileB = tileA + 8;
    const int* nA = nbrR + (size_t)(tileA*16 + row) * NTAPP;
    const int* nB = nbrR + (size_t)(tileB*16 + row) * NTAPP;

    v8f cA0 = (v8f)0.0f, cA1 = (v8f)0.0f, cB0 = (v8f)0.0f, cB1 = (v8f)0.0f;

    auto tap_body = [&](int k, int sA, int sB) {
        Frag16 aA, aB;
        if (sA >= 0) {
            const unsigned short* ar = fin + (size_t)sA * HID;
            aA.q.lo = *(const u32x4*)(ar + kh*8);
            aA.q.hi = *(const u32x4*)(ar + 16 + kh*8);
        } else { aA.q.lo = (u32x4){0,0,0,0}; aA.q.hi = (u32x4){0,0,0,0}; }
        if (sB >= 0) {
            const unsigned short* ar = fin + (size_t)sB * HID;
            aB.q.lo = *(const u32x4*)(ar + kh*8);
            aB.q.hi = *(const u32x4*)(ar + 16 + kh*8);
        } else { aB.q.lo = (u32x4){0,0,0,0}; aB.q.hi = (u32x4){0,0,0,0}; }

        const unsigned short* wb = sw + k*1024;       // LDS: ds_load_b128 x4
        Frag16 b0, b1;
        b0.q.lo = *(const u32x4*)(wb + lane*16);
        b0.q.hi = *(const u32x4*)(wb + lane*16 + 8);
        b1.q.lo = *(const u32x4*)(wb + 512 + lane*16);
        b1.q.hi = *(const u32x4*)(wb + 512 + lane*16 + 8);

        // 4 independent accumulator chains -> no back-to-back WMMA hazards
        cA0 = __builtin_amdgcn_wmma_f32_16x16x32_bf16(false, aA.v, false, b0.v, (short)0, cA0, false, false);
        cB0 = __builtin_amdgcn_wmma_f32_16x16x32_bf16(false, aB.v, false, b0.v, (short)0, cB0, false, false);
        cA1 = __builtin_amdgcn_wmma_f32_16x16x32_bf16(false, aA.v, false, b1.v, (short)0, cA1, false, false);
        cB1 = __builtin_amdgcn_wmma_f32_16x16x32_bf16(false, aB.v, false, b1.v, (short)0, cB1, false, false);
    };

    for (int k0 = 0; k0 < 24; k0 += 4) {              // 6 groups of 4 taps
        u32x4 iA = *(const u32x4*)(nA + k0);          // one b128 = 4 tap indices
        u32x4 iB = *(const u32x4*)(nB + k0);
        #pragma unroll
        for (int j = 0; j < 4; ++j)
            tap_body(k0 + j, (int)iA[j], (int)iB[j]);
    }
    {                                                 // tail: taps 24..26 (27 is pad)
        u32x4 iA = *(const u32x4*)(nA + 24);
        u32x4 iB = *(const u32x4*)(nB + 24);
        #pragma unroll
        for (int j = 0; j < 3; ++j)
            tap_body(24 + j, (int)iA[j], (int)iB[j]);
    }

    epi_pair(tileA, lane, cA0, cA1, bias, res1, res2, fout, outStride);
    epi_pair(tileB, lane, cB0, cB1, bias, res1, res2, fout, outStride);
}

// mlp_2: out = cat[M,256] @ w2 + b2 -> f32 [M,128]; w2 fragments in LDS (64 KB);
// 8 independent accumulator chains; float4 (b128) output stores.
__global__ void __launch_bounds__(256)
k_final_wmma(const unsigned short* __restrict__ cat, const unsigned short* __restrict__ wpack,
             const float* __restrict__ b2, float* __restrict__ out) {
    __shared__ unsigned short sw[64*512];             // 8 K-chunks x 8 N-tiles x 512 halfs
    {
        u32x4*       d = (u32x4*)sw;
        const u32x4* s = (const u32x4*)wpack;
        for (int i = threadIdx.x; i < 64*64; i += 256) d[i] = s[i];
    }
    __syncthreads();

    int waveid = threadIdx.x >> 5;
    int lane   = threadIdx.x & 31;
    int row = lane & 15, kh = lane >> 4;
    int tile = blockIdx.x*8 + waveid;
    const unsigned short* arow = cat + (size_t)(tile*16 + row) * CATDIM;

    v8f c[8];
    #pragma unroll
    for (int t = 0; t < 8; ++t) c[t] = (v8f)0.0f;

    for (int kk = 0; kk < 8; ++kk) {
        Frag16 a;
        a.q.lo = *(const u32x4*)(arow + kk*32 + kh*8);
        a.q.hi = *(const u32x4*)(arow + kk*32 + 16 + kh*8);
        #pragma unroll
        for (int t = 0; t < 8; ++t) {
            const unsigned short* wb = sw + (size_t)(kk*8 + t)*512 + lane*16;
            Frag16 b;
            b.q.lo = *(const u32x4*)(wb);
            b.q.hi = *(const u32x4*)(wb + 8);
            c[t] = __builtin_amdgcn_wmma_f32_16x16x32_bf16(false, a.v, false, b.v, (short)0, c[t], false, false);
        }
    }

    // tiles 4u..4u+3 hold columns u*64 + 4*(lane&15) + {0..3} -> float4 stores
    int c4 = 4*(lane & 15);
    #pragma unroll
    for (int u = 0; u < 2; ++u) {
        f32x4 bb = *(const f32x4*)(b2 + u*64 + c4);
        #pragma unroll
        for (int r = 0; r < 8; ++r) {
            f32x4 v;
            v.x = c[4*u+0][r] + bb.x;
            v.y = c[4*u+1][r] + bb.y;
            v.z = c[4*u+2][r] + bb.z;
            v.w = c[4*u+3][r] + bb.w;
            *(f32x4*)(out + (size_t)(tile*16 + r + kh*8)*OUTDIM + u*64 + c4) = v;
        }
    }
}

// ---------------- host orchestration ----------------
extern "C" void kernel_launch(void* const* d_in, const int* in_sizes, int n_in,
                              void* d_out, int out_size, void* d_ws, size_t ws_size,
                              hipStream_t stream) {
    const float* p      = (const float*)d_in[0];
    const float* w1     = (const float*)d_in[1];
    const float* b1     = (const float*)d_in[2];
    const float* w_mlp  = (const float*)d_in[3];
    const float* b_mlp  = (const float*)d_in[4];
    const float* w_conv = (const float*)d_in[5];
    const float* b_conv = (const float*)d_in[6];
    const float* w2     = (const float*)d_in[7];
    const float* b2     = (const float*)d_in[8];
    float* out = (float*)d_out;
    (void)in_sizes; (void)n_in; (void)out_size; (void)ws_size;

    char* ws = (char*)d_ws;
    size_t off = 0;
    auto alloc = [&](size_t bytes) -> char* {
        char* r = ws + off;
        off = (off + bytes + 255) & ~(size_t)255;
        return r;
    };
    unsigned short* pts = (unsigned short*)alloc((size_t)MPTS*HID*2);     // 8 MB
    unsigned short* h0  = (unsigned short*)alloc((size_t)MPTS*HID*2);     // 8 MB
    unsigned short* h1  = (unsigned short*)alloc((size_t)MPTS*HID*2);     // 8 MB
    unsigned short* cat = (unsigned short*)alloc((size_t)MPTS*CATDIM*2);  // 64 MB
    int* nbrR   = (int*)alloc((size_t)MPTS*NTAPP*4);                      // 14.7 MB
    int* gcpk   = (int*)alloc((size_t)MPTS*4);
    int* grid   = (int*)alloc((size_t)BATCH*101*101*101*4);               // 8.2 MB (max D)
    int* minbuf = (int*)alloc(64);
    unsigned short* wpc = (unsigned short*)alloc((size_t)432*1024*2);     // conv weights
    unsigned short* wpm = (unsigned short*)alloc((size_t)8*1024*2);       // mlp weights
    unsigned short* wp2 = (unsigned short*)alloc((size_t)64*512*2);       // w2

    static const float gs[NSCALE] = {0.01f,0.02f,0.04f,0.08f,0.16f,0.32f,0.64f,1.28f};
    static const int   Ds[NSCALE] = {101,51,26,14,8,5,3,2};               // ceil(1/g)+1

    // weight repacks + input MLP (cheap, run every call; deterministic)
    k_repack32 <<<(432*1024 + 255)/256, 256, 0, stream>>>(w_conv, wpc, 432);
    k_repack32 <<<(8*1024   + 255)/256, 256, 0, stream>>>(w_mlp,  wpm, 8);
    k_repack_w2<<<(64*512   + 255)/256, 256, 0, stream>>>(w2, wp2);
    k_pts      <<<(MPTS     + 255)/256, 256, 0, stream>>>(p, w1, b1, pts);

    dim3 mlpGrid(MPTS/32/4), mlpBlk(128);    // 4 waves/block, 2 tiles per wave
    dim3 convGrid(MPTS/16/16), convBlk(256); // 8 waves/block, 2 tiles per wave
    dim3 finGrid(MPTS/16/8),  finBlk(256);   // 8 waves/block, 1 tile per wave

    for (int i = 0; i < NSCALE; ++i) {
        int D = Ds[i];
        int gridN = BATCH*D*D*D;
        k_fill_i32<<<(gridN + 255)/256, 256, 0, stream>>>(grid, -1, gridN);
        k_fill_i32<<<1, 32, 0, stream>>>(minbuf, 0x7fffffff, 8);
        k_gmin<<<(MPTS + 255)/256, 256, 0, stream>>>(p, gs[i], minbuf);
        k_hash<<<(MPTS + 255)/256, 256, 0, stream>>>(p, gs[i], minbuf, D, grid, gcpk);
        k_nbr <<<(MPTS + 255)/256, 256, 0, stream>>>(gcpk, grid, D, nbrR);

        k_mlp_wmma<<<mlpGrid, mlpBlk, 0, stream>>>(pts, wpm + (size_t)i*1024, b_mlp + i*HID, h0);
        // h1 = conv1(h0) + h0
        k_conv_wmma<<<convGrid, convBlk, 0, stream>>>(h0, nbrR, wpc + (size_t)(2*i)*NTAP*1024,
                                                      b_conv + (2*i)*HID, h0, nullptr, h1, HID);
        // cat[:, i*32:(i+1)*32] = conv2(h1) + h1 + h0   (== h2 + h0)
        k_conv_wmma<<<convGrid, convBlk, 0, stream>>>(h1, nbrR, wpc + (size_t)(2*i+1)*NTAP*1024,
                                                      b_conv + (2*i+1)*HID, h1, h0,
                                                      cat + i*HID, CATDIM);
    }

    k_final_wmma<<<finGrid, finBlk, 0, stream>>>(cat, wp2, b2, out);
}
